// AttentiveStatisticsPooling_7456063226188
// MI455X (gfx1250) — matrix operand order repaired and checked by reading, not compiled, not run on hardware
//
#include <hip/hip_runtime.h>
#include <hip/hip_bf16.h>
#include <math.h>

typedef _Float16 f16;
typedef __attribute__((ext_vector_type(16))) _Float16 v16h;
typedef __attribute__((ext_vector_type(8)))  _Float16 v8h;
typedef __attribute__((ext_vector_type(2)))  _Float16 v2h;
typedef __attribute__((ext_vector_type(8)))  float    v8f;

constexpr int N  = 16;
constexpr int C  = 512;
constexpr int L  = 3000;
constexpr int A  = 128;
constexpr int C3 = 1536;
constexpr float EPS_STD = 1e-12f;
constexpr float BN_EPS  = 1e-5f;

__device__ __forceinline__ int valid_len(float rel) {
  int v = (int)ceilf(rel * (float)L);
  if (v < 1) v = 1;
  if (v > L) v = L;
  return v;
}

// A-fragment for v_wmma_f32_16x16x32_f16 (16x32 f16, MxK):
// lane m = lane&15 selects row M; g = lane>>4 adds K offset 16.
// VGPR v holds K pair { 2v + 16g , +1 } (contiguous pair).
// `row` must already point at  base + M*ld + kbase + 16*g.
// (Pairs {0..15} are contiguous: compiler merges into 2x b128.)
__device__ __forceinline__ v16h load_afrag(const f16* __restrict__ row) {
  v16h a;
#pragma unroll
  for (int v = 0; v < 8; ++v) {
    const int off = 2 * v;
    v2h p = *(const v2h*)(row + off);
    a[2 * v]     = p.x;
    a[2 * v + 1] = p.y;
  }
  return a;
}

// ---------------------------------------------------------------------------
// Kernel 0: masked global mean/std per (n,c).  std = sqrt(max(m - m^2, eps))
// because the reference's var_like = sum(w*(x - mean^2)) with sum(w)=1.
// ---------------------------------------------------------------------------
__global__ __launch_bounds__(256) void k_stats(const float* __restrict__ x,
                                               const float* __restrict__ lengths,
                                               float* __restrict__ mean_g,
                                               float* __restrict__ std_g) {
  const int nc = blockIdx.x;              // 0 .. N*C-1
  const int n  = nc / C;
  const int valid = valid_len(lengths[n]);
  const float* row = x + (size_t)nc * L;
  float s = 0.0f;
  for (int l = threadIdx.x; l < valid; l += 256) s += row[l];
  __shared__ float red[256];
  red[threadIdx.x] = s;
  __syncthreads();
  for (int st = 128; st > 0; st >>= 1) {
    if (threadIdx.x < st) red[threadIdx.x] += red[threadIdx.x + st];
    __syncthreads();
  }
  if (threadIdx.x == 0) {
    const float m = red[0] / (float)valid;
    mean_g[nc] = m;
    std_g[nc]  = sqrtf(fmaxf(m - m * m, EPS_STD));
  }
}

// ---------------------------------------------------------------------------
// Kernel 1: convert W1[:, :C] and W2 to f16; precompute BN affine s,t.
// ---------------------------------------------------------------------------
__global__ __launch_bounds__(256) void k_wconv(const float* __restrict__ W1,
                                               const float* __restrict__ W2,
                                               const float* __restrict__ gamma,
                                               const float* __restrict__ beta,
                                               const float* __restrict__ mu,
                                               const float* __restrict__ var,
                                               f16* __restrict__ W1h,
                                               f16* __restrict__ W2h,
                                               float* __restrict__ sA,
                                               float* __restrict__ tA) {
  const int i = blockIdx.x * 256 + threadIdx.x;   // 0 .. 65535
  if (i < A * C) {
    const int a = i >> 9, c = i & (C - 1);
    W1h[i] = (f16)W1[(size_t)a * C3 + c];
  }
  if (i < C * A) W2h[i] = (f16)W2[i];
  if (i < A) {
    const float s = gamma[i] * rsqrtf(var[i] + BN_EPS);
    sA[i] = s;
    tA[i] = beta[i] - mu[i] * s;
  }
}

// ---------------------------------------------------------------------------
// Kernel 2: bias2[n,a] = b1[a] + W1[a,C: 2C]·mean_g[n] + W1[a,2C:3C]·std_g[n]
// ---------------------------------------------------------------------------
__global__ __launch_bounds__(128) void k_bias2(const float* __restrict__ W1,
                                               const float* __restrict__ b1,
                                               const float* __restrict__ mean_g,
                                               const float* __restrict__ std_g,
                                               float* __restrict__ bias2) {
  const int na = blockIdx.x;            // 0 .. N*A-1
  const int n = na / A, a = na % A;
  const float* wm = W1 + (size_t)a * C3 + C;
  const float* ws = W1 + (size_t)a * C3 + 2 * C;
  const float* mg = mean_g + (size_t)n * C;
  const float* sg = std_g + (size_t)n * C;
  float s = 0.0f;
  for (int c = threadIdx.x; c < C; c += 128) s += wm[c] * mg[c] + ws[c] * sg[c];
  __shared__ float red[128];
  red[threadIdx.x] = s;
  __syncthreads();
  for (int st = 64; st > 0; st >>= 1) {
    if (threadIdx.x < st) red[threadIdx.x] += red[threadIdx.x + st];
    __syncthreads();
  }
  if (threadIdx.x == 0) bias2[na] = red[0] + b1[a];
}

// ---------------------------------------------------------------------------
// Kernel 3: LDS-tiled transpose + f32->f16 convert: x[N,C,L] -> xT[N,L,C]
// so WMMA B-fragments (K along C) become contiguous 32B vector loads.
// ---------------------------------------------------------------------------
__global__ __launch_bounds__(256) void k_transpose(const float* __restrict__ x,
                                                   f16* __restrict__ xT) {
  __shared__ float tile[32][33];
  const int n  = blockIdx.z;
  const int l0 = blockIdx.x * 32;
  const int c0 = blockIdx.y * 32;
  const int tx = threadIdx.x, ty = threadIdx.y;   // 32 x 8
#pragma unroll
  for (int k = 0; k < 4; ++k) {
    const int c = c0 + ty + 8 * k;
    const int l = l0 + tx;
    tile[ty + 8 * k][tx] = (l < L) ? x[((size_t)n * C + c) * L + l] : 0.0f;
  }
  __syncthreads();
#pragma unroll
  for (int k = 0; k < 4; ++k) {
    const int l = l0 + ty + 8 * k;
    const int c = c0 + tx;
    if (l < L) xT[((size_t)n * L + l) * C + c] = (f16)tile[tx][ty + 8 * k];
  }
}

// ---------------------------------------------------------------------------
// Kernel 4: h = tanh(BN(relu(W1a·x + bias2)))  via f16 WMMA, f32 accum.
// Block: 8 waves, wave w owns rows a0=16w, 64 L-columns (4 sub-tiles).
// Out-of-range l lanes load a *clamped* valid row unconditionally: WMMA
// columns are independent and those columns are never stored, so no zero-fill
// or EXEC branching is needed in the hot loop.
// Output hT[N,L,A] f16 — D-fragment packs 8 consecutive `a` per lane.
// ---------------------------------------------------------------------------
__global__ __launch_bounds__(256) void k_h_gemm(const f16* __restrict__ W1h,
                                                const f16* __restrict__ xT,
                                                const float* __restrict__ bias2,
                                                const float* __restrict__ sA,
                                                const float* __restrict__ tA,
                                                f16* __restrict__ hT) {
  const int n  = blockIdx.y;
  const int l0 = blockIdx.x * 64;
  const int wave = threadIdx.x >> 5, lane = threadIdx.x & 31;
  const int a0 = wave * 16, m = lane & 15, g = lane >> 4;

  // Per-lane clamped B row pointers for the 4 L sub-tiles.
  const f16* brow[4];
#pragma unroll
  for (int j = 0; j < 4; ++j) {
    int l = l0 + 16 * j + m;
    int lc = (l < L) ? l : (L - 1);
    brow[j] = xT + ((size_t)n * L + lc) * C + 16 * g;
  }

  v8f acc[4];
#pragma unroll
  for (int j = 0; j < 4; ++j)
#pragma unroll
    for (int r = 0; r < 8; ++r) acc[j][r] = 0.0f;

  const f16* wbase = W1h + (size_t)(a0 + m) * C + 16 * g;
  for (int ks = 0; ks < 16; ++ks) {          // K = C = 512, step 32
    const int cb = ks * 32;
    const v16h Af = load_afrag(wbase + cb);
#pragma unroll
    for (int j = 0; j < 4; ++j) {
      const v16h Bf = *(const v16h*)(brow[j] + cb);
      acc[j] = __builtin_amdgcn_wmma_f32_16x16x32_f16(
          false, Af, false, Bf, (short)0, acc[j], false, false);
    }
  }

  // Hoist per-row epilogue constants (a fixed per r across sub-tiles).
  float bv[8], sv[8], tv[8];
#pragma unroll
  for (int r = 0; r < 8; ++r) {
    const int a = a0 + r + 8 * g;
    bv[r] = bias2[n * A + a];
    sv[r] = sA[a];
    tv[r] = tA[a];
  }

#pragma unroll
  for (int j = 0; j < 4; ++j) {
    const int l = l0 + 16 * j + m;
    v8h hv;
#pragma unroll
    for (int r = 0; r < 8; ++r) {
      float v = acc[j][r] + bv[r];
      v = fmaxf(v, 0.0f);                    // ReLU
      v = v * sv[r] + tv[r];                 // BatchNorm affine
      hv[r] = (f16)tanhf(v);
    }
    if (l < L)
      *(v8h*)(hT + ((size_t)n * L + l) * A + a0 + 8 * g) = hv;
  }
}

// ---------------------------------------------------------------------------
// Kernel 5: scores a = W2·h (+b2) via WMMA; e = exp(a) on valid l (softmax is
// shift-invariant, |a| is small); accumulate S0 = Σe, S1 = Σe·x per (n,c)
// with lane-group shuffles + f32 global atomics.  B loads are clamped
// (columns l >= valid get weight 0, so garbage columns are harmless).
// Block: 8 waves; wave owns 16 c-rows, 128 L-columns (8 sub-tiles).
// ---------------------------------------------------------------------------
__global__ __launch_bounds__(256) void k_attn_pool(const f16* __restrict__ W2h,
                                                   const f16* __restrict__ hT,
                                                   const float* __restrict__ x,
                                                   const float* __restrict__ b2,
                                                   const float* __restrict__ lengths,
                                                   float* __restrict__ S0,
                                                   float* __restrict__ S1) {
  const int n = blockIdx.z;
  const int wave = threadIdx.x >> 5, lane = threadIdx.x & 31;
  const int c0 = blockIdx.y * 128 + wave * 16;
  const int lbase = blockIdx.x * 128;
  const int m = lane & 15, g = lane >> 4;
  const int valid = valid_len(lengths[n]);

  // Preload all 4 A-fragments (K = A = 128) and per-row constants.
  v16h Af[4];
  const f16* wbase = W2h + (size_t)(c0 + m) * A + 16 * g;
#pragma unroll
  for (int ks = 0; ks < 4; ++ks) Af[ks] = load_afrag(wbase + ks * 32);

  float b2v[8];
  const float* xrow[8];
#pragma unroll
  for (int r = 0; r < 8; ++r) {
    const int c = c0 + r + 8 * g;
    b2v[r]  = b2[c];
    xrow[r] = x + ((size_t)n * C + c) * L;
  }

  float p0[8], p1[8];
#pragma unroll
  for (int r = 0; r < 8; ++r) { p0[r] = 0.0f; p1[r] = 0.0f; }

  for (int j = 0; j < 8; ++j) {
    const int l  = lbase + 16 * j + m;
    const int lc = (l < valid) ? l : (valid - 1);   // clamp (valid >= 1)
    const bool ok = (l < valid);                    // masked l -> weight 0

    v8f acc;
#pragma unroll
    for (int r = 0; r < 8; ++r) acc[r] = 0.0f;

    const f16* hrow = hT + ((size_t)n * L + lc) * A + 16 * g;
#pragma unroll
    for (int ks = 0; ks < 4; ++ks) {
      const v16h Bf = *(const v16h*)(hrow + ks * 32);
      acc = __builtin_amdgcn_wmma_f32_16x16x32_f16(
          false, Af[ks], false, Bf, (short)0, acc, false, false);
    }

#pragma unroll
    for (int r = 0; r < 8; ++r) {
      const float e  = ok ? expf(acc[r] + b2v[r]) : 0.0f;   // branch-free select
      const float xv = xrow[r][lc];                         // clamped, always safe
      p0[r] += e;
      p1[r] += e * xv;
    }
  }

#pragma unroll
  for (int r = 0; r < 8; ++r) {
    float s0 = p0[r], s1 = p1[r];
#pragma unroll
    for (int mk = 8; mk >= 1; mk >>= 1) {    // reduce within 16-lane group
      s0 += __shfl_xor(s0, mk, 32);
      s1 += __shfl_xor(s1, mk, 32);
    }
    if (m == 0) {
      const int c = c0 + r + 8 * g;
      atomicAdd(&S0[n * C + c], s0);
      atomicAdd(&S1[n * C + c], s1);
    }
  }
}

// ---------------------------------------------------------------------------
// Kernel 6: mean = S1/S0 ; std = sqrt(max(mean - mean^2, eps)); pack output.
// ---------------------------------------------------------------------------
__global__ __launch_bounds__(256) void k_final(const float* __restrict__ S0,
                                               const float* __restrict__ S1,
                                               float* __restrict__ out) {
  const int i = blockIdx.x * 256 + threadIdx.x;
  if (i >= N * C) return;
  const int n = i / C, c = i % C;
  const float mean = S1[i] / S0[i];
  out[(size_t)n * 2 * C + c]     = mean;
  out[(size_t)n * 2 * C + C + c] = sqrtf(fmaxf(mean - mean * mean, EPS_STD));
}

// ---------------------------------------------------------------------------
extern "C" void kernel_launch(void* const* d_in, const int* in_sizes, int n_in,
                              void* d_out, int out_size, void* d_ws, size_t ws_size,
                              hipStream_t stream) {
  const float* x        = (const float*)d_in[0];
  const float* lengths  = (const float*)d_in[1];
  const float* W1       = (const float*)d_in[2];
  const float* b1       = (const float*)d_in[3];
  const float* bn_gamma = (const float*)d_in[4];
  const float* bn_beta  = (const float*)d_in[5];
  const float* bn_mean  = (const float*)d_in[6];
  const float* bn_var   = (const float*)d_in[7];
  const float* W2       = (const float*)d_in[8];
  const float* b2       = (const float*)d_in[9];
  float* out = (float*)d_out;

  // Workspace carve (all chunks 256B-aligned by construction).
  char* p = (char*)d_ws;
  float* mean_g = (float*)p;              p += (size_t)N * C * 4;   // 32 KB
  float* std_g  = (float*)p;              p += (size_t)N * C * 4;   // 32 KB
  float* bias2  = (float*)p;              p += (size_t)N * A * 4;   //  8 KB
  float* sA     = (float*)p;              p += 512;
  float* tA     = (float*)p;              p += 512;
  float* S0     = (float*)p;              p += (size_t)N * C * 4;   // 32 KB
  float* S1     = (float*)p;              p += (size_t)N * C * 4;   // 32 KB (adjacent to S0)
  f16*   W1h    = (f16*)p;                p += (size_t)A * C * 2;   // 128 KB
  f16*   W2h    = (f16*)p;                p += (size_t)C * A * 2;   // 128 KB
  f16*   xT     = (f16*)p;                p += (size_t)N * L * C * 2; // 49.2 MB
  f16*   hT     = (f16*)p;                p += (size_t)N * L * A * 2; // 12.3 MB

  // Zero the atomic accumulators (S0 and S1 are contiguous).
  hipMemsetAsync(S0, 0, (size_t)2 * N * C * sizeof(float), stream);

  k_stats<<<dim3(N * C), dim3(256), 0, stream>>>(x, lengths, mean_g, std_g);

  k_wconv<<<dim3((A * C + 255) / 256), dim3(256), 0, stream>>>(
      W1, W2, bn_gamma, bn_beta, bn_mean, bn_var, W1h, W2h, sA, tA);

  k_bias2<<<dim3(N * A), dim3(128), 0, stream>>>(W1, b1, mean_g, std_g, bias2);

  k_transpose<<<dim3((L + 31) / 32, C / 32, N), dim3(32, 8), 0, stream>>>(x, xT);

  k_h_gemm<<<dim3((L + 63) / 64, N), dim3(256), 0, stream>>>(
      W1h, xT, bias2, sA, tA, hT);

  k_attn_pool<<<dim3((L + 127) / 128, C / 128, N), dim3(256), 0, stream>>>(
      W2h, hT, x, b2, lengths, S0, S1);

  k_final<<<dim3((N * C + 255) / 256), dim3(256), 0, stream>>>(S0, S1, out);
}
